// _Expert_linear_45414984188388
// MI455X (gfx1250) — matrix operand order repaired and checked by reading, not compiled, not run on hardware
//
#include <hip/hip_runtime.h>

#define T_TOK 8192
#define D_IN  1024
#define D_OUT 1024
#define N_EXP 8

typedef __bf16 bf16_t;
typedef __attribute__((ext_vector_type(16))) __bf16 v16bf;
typedef __attribute__((ext_vector_type(8)))  __bf16 v8bf;
typedef __attribute__((ext_vector_type(8)))  float  v8f;

union frag16 { v16bf v; v8bf h[2]; };

// ---------------------------------------------------------------------------
// Pre-pass: pack two f32 -> one dword of two RNE-rounded bf16.
// 96 MB of traffic total for both passes -> ~4 us at 23.3 TB/s, amortized once.
// ---------------------------------------------------------------------------
__global__ void f32_to_bf16_pk(const float* __restrict__ src,
                               unsigned int* __restrict__ dst, int n2) {
  int i = blockIdx.x * blockDim.x + threadIdx.x;
  const int stride = gridDim.x * blockDim.x;
  for (; i < n2; i += stride) {
    unsigned long long p = ((const unsigned long long*)src)[i];
    unsigned int lo = (unsigned int)p;
    unsigned int hi = (unsigned int)(p >> 32);
    lo = (lo + 0x7fffu + ((lo >> 16) & 1u)) >> 16;   // round-to-nearest-even
    hi = (hi + 0x7fffu + ((hi >> 16) & 1u)) >> 16;
    dst[i] = lo | (hi << 16);
  }
}

// ---------------------------------------------------------------------------
// Grouped GEMM: one block = 128x128 C tile of one expert; 8 wave32s,
// each wave = 32x64 (2x4 WMMA tiles), K stepped by 32 (bf16 WMMA K).
// ---------------------------------------------------------------------------
__global__ __launch_bounds__(256) void moe_gemm_bf16(
    const bf16_t* __restrict__ A,     // [T, D_IN]      bf16
    const bf16_t* __restrict__ W,     // [E, D_OUT, D_IN] bf16
    const float*  __restrict__ bias,  // [E, D_OUT]
    const int*    __restrict__ counts,// [E]
    float* __restrict__ out)          // [T, D_OUT]
{
  const int e = blockIdx.z;
  // uniform prefix sum over <=8 counts (scalar path)
  int seg_off = 0;
  for (int i = 0; i < e; ++i) seg_off += counts[i];
  const int seg_cnt = counts[e];

  const int n0   = blockIdx.x * 128;
  const int m0   = blockIdx.y * 128;            // offset within this expert's segment
  const int lane = threadIdx.x & 31;
  const int wid  = threadIdx.x >> 5;
  const int wm   = (wid & 3) * 32;              // wave m-offset in tile
  const int wn   = (wid >> 2) * 64;             // wave n-offset in tile
  const int l15  = lane & 15;
  const int ksel = lane >> 4;                   // which K-half this lane holds

  // A fragment pointers (lanes 0-15 K=[0..15], lanes 16-31 K=[16..31])
  const bf16_t* pa[2];
#pragma unroll
  for (int mi = 0; mi < 2; ++mi) {
    int r = seg_off + m0 + wm + mi * 16 + l15;
    r = (r < T_TOK - 1) ? r : (T_TOK - 1);      // clamp (VALU select, no divergence)
    pa[mi] = A + (size_t)r * D_IN + ksel * 16;
  }
  // B fragment pointers (lane = N; lanes 0-15 K=[0..15], lanes 16-31 K=[16..31])
  const bf16_t* Wb = W + (size_t)e * D_OUT * D_IN;
  const bf16_t* pb[4];
#pragma unroll
  for (int ni = 0; ni < 4; ++ni) {
    int n = n0 + wn + ni * 16 + l15;
    pb[ni] = Wb + (size_t)n * D_IN + ksel * 16;
  }

  v8f acc[2][4] = {};

  for (int k = 0; k < D_IN; k += 32) {
    frag16 a[2], b[4];
#pragma unroll
    for (int mi = 0; mi < 2; ++mi) {
      a[mi].h[0] = *(const v8bf*)(pa[mi]);      // global_load_b128
      a[mi].h[1] = *(const v8bf*)(pa[mi] + 8);
      pa[mi] += 32;
    }
#pragma unroll
    for (int ni = 0; ni < 4; ++ni) {
      b[ni].h[0] = *(const v8bf*)(pb[ni]);
      b[ni].h[1] = *(const v8bf*)(pb[ni] + 8);
      pb[ni] += 32;
    }
#pragma unroll
    for (int mi = 0; mi < 2; ++mi)
#pragma unroll
      for (int ni = 0; ni < 4; ++ni)
        acc[mi][ni] = __builtin_amdgcn_wmma_f32_16x16x32_bf16(
            false, a[mi].v, false, b[ni].v, (short)0, acc[mi][ni], false, false);
  }

  // Epilogue: bias add + guarded f32 stores (C layout: lane half selects M+8)
  const float* be = bias + (size_t)e * D_OUT;
#pragma unroll
  for (int ni = 0; ni < 4; ++ni) {
    const int col = n0 + wn + ni * 16 + l15;
    const float bv = be[col];
#pragma unroll
    for (int mi = 0; mi < 2; ++mi) {
      v8f r = acc[mi][ni];
#pragma unroll
      for (int v = 0; v < 8; ++v) {
        const int rowLocal = m0 + wm + mi * 16 + v + ksel * 8;
        const int row = seg_off + rowLocal;
        if (rowLocal < seg_cnt && row < T_TOK)
          out[(size_t)row * D_OUT + col] = r[v] + bv;
      }
    }
  }
}

// ---------------------------------------------------------------------------
extern "C" void kernel_launch(void* const* d_in, const int* in_sizes, int n_in,
                              void* d_out, int out_size, void* d_ws, size_t ws_size,
                              hipStream_t stream) {
  const float* inp    = (const float*)d_in[0];
  const float* weight = (const float*)d_in[1];
  const float* bias   = (const float*)d_in[2];
  const int*   counts = (const int*)d_in[3];
  float* out = (float*)d_out;

  const int nInp = in_sizes[0];              // T * D_IN
  const int nW   = in_sizes[1];              // E * D_OUT * D_IN

  unsigned int* wsA = (unsigned int*)d_ws;   // bf16 inp  (16 MB)
  unsigned int* wsW = wsA + (nInp >> 1);     // bf16 weight (16 MB)

  f32_to_bf16_pk<<<2048, 256, 0, stream>>>(inp,    wsA, nInp >> 1);
  f32_to_bf16_pk<<<2048, 256, 0, stream>>>(weight, wsW, nW  >> 1);

  // 8 n-tiles x 8 m-tiles(per expert segment) x 8 experts
  dim3 grid(D_OUT / 128, (T_TOK / N_EXP) / 128, N_EXP);
  moe_gemm_bf16<<<grid, 256, 0, stream>>>((const bf16_t*)wsA, (const bf16_t*)wsW,
                                          bias, counts, out);
}
